// NMT_17102559773210
// MI455X (gfx1250) — matrix operand order repaired and checked by reading, not compiled, not run on hardware
//
#include <hip/hip_runtime.h>
#include <hip/hip_bf16.h>

// ---------------------------------------------------------------------------
// Types for WMMA fragments (wave32, gfx1250)
// ---------------------------------------------------------------------------
typedef __attribute__((ext_vector_type(16))) _Float16 v16h;
typedef __attribute__((ext_vector_type(8)))  _Float16 v8h;
typedef __attribute__((ext_vector_type(8)))  float    v8f;

#define Bn 32
#define Sn 64
#define Tn 64
#define En 512
#define Hn 512
#define Vt 32000

// ---------------------------------------------------------------------------
// Generic wave-tiled GEMM:  C[M,N] = A[M,K] (f16) * W[N,K]^T (f16) + bias
// One wave computes one 16x16 tile via v_wmma_f32_16x16x32_f16 over K.
// Optional tanh activation; optional f32 and/or f16 outputs.
// ---------------------------------------------------------------------------
__global__ __launch_bounds__(256)
void k_gemm_f16(const _Float16* __restrict__ A, const _Float16* __restrict__ W,
                const float* __restrict__ bias, float* __restrict__ C,
                _Float16* __restrict__ Ch, int M, int N, int K, int act)
{
    const int lane  = threadIdx.x & 31;
    const int wave  = threadIdx.x >> 5;
    const int tilesN = N >> 4;
    const int tile  = blockIdx.x * 8 + wave;
    const int tM = tile / tilesN;
    const int tN = tile - tM * tilesN;
    if (tM * 16 >= M) return;                  // uniform per-wave guard

    const int row   = lane & 15;               // A row / B col within tile
    const int kA    = (lane < 16) ? 0 : 16;    // contiguous k = 16h + s (matches B)
    const int kB    = (lane < 16) ? 0 : 16;    // contiguous k = 16h + s (matches A)
    const _Float16* ap = A + (size_t)(tM * 16 + row) * K + kA;
    const _Float16* wp = W + (size_t)(tN * 16 + row) * K + kB;

    v8f acc = {};
    for (int k0 = 0; k0 < K; k0 += 32) {
        __builtin_prefetch(wp + k0 + 128, 0, 1);   // global_prefetch_b8
        v16h a;
        const v8h a0 = *(const v8h*)(ap + k0);
        const v8h a1 = *(const v8h*)(ap + k0 + 8);
#pragma unroll
        for (int i = 0; i < 8; ++i) { a[i] = a0[i]; a[i + 8] = a1[i]; }
        const v16h b = *(const v16h*)(wp + k0);
        acc = __builtin_amdgcn_wmma_f32_16x16x32_f16(
            false, a, false, b, (short)0, acc, false, false);
    }

    const int cn = tN * 16 + (lane & 15);
    const float bi = bias ? bias[cn] : 0.f;
    const int mhalf = (lane < 16) ? 0 : 8;
#pragma unroll
    for (int r = 0; r < 8; ++r) {
        const int cm = tM * 16 + r + mhalf;
        float v = acc[r] + bi;
        if (act == 1) v = tanhf(v);
        if (C)  C[(size_t)cm * N + cn]  = v;
        if (Ch) Ch[(size_t)cm * N + cn] = (_Float16)v;
    }
}

// ---------------------------------------------------------------------------
// Elementwise / gather / conversion kernels
// ---------------------------------------------------------------------------
__global__ void k_zero_f(float* p, int n)
{ int i = blockIdx.x * blockDim.x + threadIdx.x; if (i < n) p[i] = 0.f; }

__global__ void k_zero_h(_Float16* p, int n)
{ int i = blockIdx.x * blockDim.x + threadIdx.x; if (i < n) p[i] = (_Float16)0.f; }

__global__ void k_cvt_h(const float* __restrict__ s, _Float16* __restrict__ d, int n)
{ int i = blockIdx.x * blockDim.x + threadIdx.x; if (i < n) d[i] = (_Float16)s[i]; }

// src_x[r, :] = src_emb[src_tokens_flat[r]]  (r = 0..S*B-1, reshape semantics)
__global__ void k_embed_src(const int* __restrict__ tok, const float* __restrict__ emb,
                            _Float16* __restrict__ out)
{
    int i = blockIdx.x * blockDim.x + threadIdx.x;
    if (i >= Sn * Bn * En) return;
    int r = i >> 9, e = i & 511;
    out[i] = (_Float16)emb[(size_t)tok[r] * En + e];
}

// tgt_x[t*B+b, :] = tgt_emb[tgt[b, t]]   (transpose(1,0,2) semantics)
__global__ void k_embed_tgt(const int* __restrict__ tok, const float* __restrict__ emb,
                            _Float16* __restrict__ out)
{
    int i = blockIdx.x * blockDim.x + threadIdx.x;
    if (i >= Tn * Bn * En) return;
    int r = i >> 9, e = i & 511;
    int t = r >> 5, b = r & 31;
    out[i] = (_Float16)emb[(size_t)tok[b * Tn + t] * En + e];
}

// Wcomb[n, 0:1024] = dec_Wih[n,:]; Wcomb[n, 1024:1536] = dec_Whh[n,:]
__global__ void k_build_wcomb(const float* __restrict__ Wih, const float* __restrict__ Whh,
                              _Float16* __restrict__ d)
{
    int i = blockIdx.x * blockDim.x + threadIdx.x;
    if (i >= 2048 * 1536) return;
    int n = i / 1536, k = i - n * 1536;
    float v = (k < 1024) ? Wih[(size_t)n * 1024 + k] : Whh[(size_t)n * 512 + (k - 1024)];
    d[i] = (_Float16)v;
}

// Encoder LSTM cell with length masking (i,f,g,o gate order)
__global__ void k_enc_cell(const float* __restrict__ Xg, const float* __restrict__ Hg,
                           const float* __restrict__ bias, const int* __restrict__ lens,
                           int t, float* __restrict__ h, float* __restrict__ c,
                           _Float16* __restrict__ h_h, float* __restrict__ out)
{
    int idx = blockIdx.x * blockDim.x + threadIdx.x;
    if (idx >= Bn * Hn) return;
    int b = idx >> 9, j = idx & 511;
    const float* xg = Xg + (size_t)(t * Bn + b) * 2048;
    const float* hg = Hg + (size_t)b * 2048;
    float gi = xg[j]        + hg[j]        + bias[j];
    float gf = xg[j + 512]  + hg[j + 512]  + bias[j + 512];
    float gg = xg[j + 1024] + hg[j + 1024] + bias[j + 1024];
    float go = xg[j + 1536] + hg[j + 1536] + bias[j + 1536];
    float si = 1.f / (1.f + __expf(-gi));
    float sf = 1.f / (1.f + __expf(-gf));
    float so = 1.f / (1.f + __expf(-go));
    float cn = sf * c[idx] + si * tanhf(gg);
    float hn = so * tanhf(cn);
    float valid = (t < lens[b]) ? 1.f : 0.f;
    float hv = valid * hn + (1.f - valid) * h[idx];
    float cv = valid * cn + (1.f - valid) * c[idx];
    h[idx] = hv; c[idx] = cv; h_h[idx] = (_Float16)hv;
    out[(size_t)t * Bn * Hn + idx] = valid * hv;
}

// Decoder LSTM cell (gates pre-summed; no masking)
__global__ void k_dec_cell(const float* __restrict__ g, float* __restrict__ h,
                           float* __restrict__ c, _Float16* __restrict__ h_h)
{
    int idx = blockIdx.x * blockDim.x + threadIdx.x;
    if (idx >= Bn * Hn) return;
    int b = idx >> 9, j = idx & 511;
    const float* gb = g + (size_t)b * 2048;
    float gi = gb[j], gf = gb[j + 512], gg = gb[j + 1024], go = gb[j + 1536];
    float si = 1.f / (1.f + __expf(-gi));
    float sf = 1.f / (1.f + __expf(-gf));
    float so = 1.f / (1.f + __expf(-go));
    float cn = sf * c[idx] + si * tanhf(gg);
    float hn = so * tanhf(cn);
    c[idx] = cn; h[idx] = hn; h_h[idx] = (_Float16)hn;
}

// src_enc[(b*S+s), 0:512] = out_f[s,b,:]; [512:1024] = out_b[s,b,:]
__global__ void k_build_srcenc(const float* __restrict__ of, const float* __restrict__ ob,
                               float* __restrict__ enc, _Float16* __restrict__ ench)
{
    int i = blockIdx.x * blockDim.x + threadIdx.x;
    if (i >= Bn * Sn * 1024) return;
    int r = i >> 10, j = i & 1023;
    int b = r >> 6, s = r & 63;
    float v = (j < 512) ? of[(size_t)(s * Bn + b) * Hn + j]
                        : ob[(size_t)(s * Bn + b) * Hn + (j - 512)];
    enc[i] = v; ench[i] = (_Float16)v;
}

// cinit[b, 0:512] = c_f[b]; [512:1024] = c_b[b]
__global__ void k_build_cinit(const float* __restrict__ cf, const float* __restrict__ cb,
                              _Float16* __restrict__ d)
{
    int i = blockIdx.x * blockDim.x + threadIdx.x;
    if (i >= Bn * 1024) return;
    int b = i >> 10, j = i & 1023;
    d[i] = (_Float16)((j < 512) ? cf[b * 512 + j] : cb[b * 512 + (j - 512)]);
}

__global__ void k_h0(const float* __restrict__ c0, float* __restrict__ h,
                     _Float16* __restrict__ h_h)
{
    int i = blockIdx.x * blockDim.x + threadIdx.x;
    if (i >= Bn * Hn) return;
    float v = tanhf(c0[i]);
    h[i] = v; h_h[i] = (_Float16)v;
}

// xcat[b] = [ y_t(b) | att_prev(b) | h(b) ]   (all f16, 1536 wide)
__global__ void k_build_xcat(const _Float16* __restrict__ tgtX, int t,
                             const _Float16* __restrict__ attPrev,
                             const _Float16* __restrict__ h_h,
                             _Float16* __restrict__ xcat)
{
    int i = blockIdx.x * blockDim.x + threadIdx.x;
    if (i >= Bn * 1536) return;
    int b = i / 1536, k = i - b * 1536;
    _Float16 v;
    if (k < 512)       v = tgtX[(size_t)(t * Bn + b) * En + k];
    else if (k < 1024) v = attPrev[(size_t)b * Hn + (k - 512)];
    else               v = h_h[(size_t)b * Hn + (k - 1024)];
    xcat[i] = v;
}

// Attention: score -> masked softmax -> context; writes xav = [h | ctx] (f16)
__global__ __launch_bounds__(256)
void k_attn(const float* __restrict__ srcAtt,   // [B*S, 512]
            const float* __restrict__ srcEnc,   // [B*S, 1024]
            const float* __restrict__ h,        // [B, 512]
            const _Float16* __restrict__ h_h,
            const int* __restrict__ lens,
            _Float16* __restrict__ xav)         // [B, 1536]
{
    __shared__ float sc[Sn];
    __shared__ float red[256];
    const int b = blockIdx.x;
    const int tid = threadIdx.x;
    // dot(src_att[b,s,:], h[b,:]) : 4 threads per s, 128 elems each
    {
        int s = tid & 63, chunk = tid >> 6;
        const float* av = srcAtt + (size_t)(b * Sn + s) * Hn + chunk * 128;
        const float* hv = h + (size_t)b * Hn + chunk * 128;
        float p = 0.f;
        for (int i = 0; i < 128; ++i) p += av[i] * hv[i];
        red[tid] = p;
    }
    __syncthreads();
    if (tid < Sn) {
        float sum = red[tid] + red[tid + 64] + red[tid + 128] + red[tid + 192];
        sc[tid] = (tid < lens[b]) ? sum : -INFINITY;
    }
    __syncthreads();
    if (tid == 0) {
        float m = -INFINITY;
        for (int i = 0; i < Sn; ++i) m = fmaxf(m, sc[i]);
        float ss = 0.f;
        for (int i = 0; i < Sn; ++i) { float e = __expf(sc[i] - m); sc[i] = e; ss += e; }
        float inv = 1.f / ss;
        for (int i = 0; i < Sn; ++i) sc[i] *= inv;
    }
    __syncthreads();
    // ctx over 1024 dims; write as f16 into xav[512:1536]
    for (int d = tid; d < 1024; d += 256) {
        float acc = 0.f;
        const float* ep = srcEnc + (size_t)(b * Sn) * 1024 + d;
        for (int s = 0; s < Sn; ++s) acc += sc[s] * ep[(size_t)s * 1024];
        xav[(size_t)b * 1536 + 512 + d] = (_Float16)acc;
    }
    for (int d = tid; d < 512; d += 256)
        xav[(size_t)b * 1536 + d] = h_h[(size_t)b * Hn + d];
}

// ---------------------------------------------------------------------------
// Fused readout: WMMA logits tiles + online logsumexp + gold gather.
// Block = one 16-row M-tile of att_ves (row = t*B+b); 8 waves stream the
// 2000 vocab tiles; cross-lane reductions via shfl_xor within 16-lane halves.
// ---------------------------------------------------------------------------
__global__ __launch_bounds__(256)
void k_readout(const _Float16* __restrict__ Ah,  // [T*B, 512]
               const _Float16* __restrict__ Wr,  // [V, 512]
               const int* __restrict__ tgt,      // [B, T]
               float* __restrict__ out)          // [T]
{
    __shared__ float lmx[8][16];
    __shared__ float lsm[8][16];
    __shared__ float lgold[16];
    const int lane = threadIdx.x & 31;
    const int wave = threadIdx.x >> 5;
    const int mt = blockIdx.x;                   // 0..127
    if (threadIdx.x < 16) lgold[threadIdx.x] = 0.f;
    __syncthreads();

    const int row   = lane & 15;
    const int kA    = (lane < 16) ? 0 : 16;
    const int kB    = (lane < 16) ? 0 : 16;
    const int half8 = (lane < 16) ? 0 : 8;
    const _Float16* ap = Ah + (size_t)(mt * 16 + row) * Hn + kA;

    // Preload all 16 K-chunks of A (constant across vocab tiles)
    v16h afrag[16];
#pragma unroll
    for (int kc = 0; kc < 16; ++kc) {
        const v8h a0 = *(const v8h*)(ap + kc * 32);
        const v8h a1 = *(const v8h*)(ap + kc * 32 + 8);
#pragma unroll
        for (int i = 0; i < 8; ++i) { afrag[kc][i] = a0[i]; afrag[kc][i + 8] = a1[i]; }
    }

    float mx[8], sm[8];
    int gidx[8];
#pragma unroll
    for (int r = 0; r < 8; ++r) {
        mx[r] = -INFINITY; sm[r] = 0.f;
        int rg = mt * 16 + r + half8;            // = t*B + b
        int t = rg >> 5, b = rg & 31;
        gidx[r] = tgt[b * Tn + t];
    }

    for (int nt = wave; nt < (Vt / 16); nt += 8) {
        const _Float16* wp = Wr + (size_t)(nt * 16 + row) * Hn + kB;
        v8f acc = {};
#pragma unroll
        for (int k0 = 0; k0 < Hn; k0 += 32) {
            __builtin_prefetch(wp + k0 + 256, 0, 1);
            const v16h bfrag = *(const v16h*)(wp + k0);
            acc = __builtin_amdgcn_wmma_f32_16x16x32_f16(
                false, afrag[k0 >> 5], false, bfrag, (short)0, acc, false, false);
        }
#pragma unroll
        for (int r = 0; r < 8; ++r) {
            float v = acc[r];
            float m = v;
            for (int off = 1; off < 16; off <<= 1)
                m = fmaxf(m, __shfl_xor(m, off, 32));
            float e = __expf(v - m);
            float s = e;
            for (int off = 1; off < 16; off <<= 1)
                s += __shfl_xor(s, off, 32);
            float nm = fmaxf(mx[r], m);
            sm[r] = sm[r] * __expf(mx[r] - nm) + s * __expf(m - nm);
            mx[r] = nm;
            int g = gidx[r];
            if (((g >> 4) == nt) && ((g & 15) == (lane & 15)))
                lgold[r + half8] = v;            // single writer across block
        }
    }

    if ((lane & 15) == 0) {
#pragma unroll
        for (int r = 0; r < 8; ++r) {
            lmx[wave][r + half8] = mx[r];
            lsm[wave][r + half8] = sm[r];
        }
    }
    __syncthreads();

    if (threadIdx.x < 16) {
        int rl = threadIdx.x;
        float M = -INFINITY;
        for (int w = 0; w < 8; ++w) M = fmaxf(M, lmx[w][rl]);
        float S = 0.f;
        for (int w = 0; w < 8; ++w) S += lsm[w][rl] * __expf(lmx[w][rl] - M);
        float lse = M + __logf(S);
        int rg = mt * 16 + rl;
        int t = rg >> 5, b = rg & 31;
        int g = tgt[b * Tn + t];
        if (g != 0) atomicAdd(&out[t], lgold[rl] - lse);
    }
}

// ---------------------------------------------------------------------------
// Host orchestration
// ---------------------------------------------------------------------------
extern "C" void kernel_launch(void* const* d_in, const int* in_sizes, int n_in,
                              void* d_out, int out_size, void* d_ws, size_t ws_size,
                              hipStream_t stream)
{
    (void)in_sizes; (void)n_in; (void)out_size; (void)ws_size;
    const int*   srcTok  = (const int*)d_in[0];
    const int*   tgtTok  = (const int*)d_in[1];
    const int*   lens    = (const int*)d_in[2];
    const float* srcEmb  = (const float*)d_in[3];
    const float* tgtEmb  = (const float*)d_in[4];
    const float* WihF    = (const float*)d_in[5];
    const float* WhhF    = (const float*)d_in[6];
    const float* bF      = (const float*)d_in[7];
    const float* WihB    = (const float*)d_in[8];
    const float* WhhB    = (const float*)d_in[9];
    const float* bB      = (const float*)d_in[10];
    const float* decWih  = (const float*)d_in[11];
    const float* decWhh  = (const float*)d_in[12];
    const float* decB    = (const float*)d_in[13];
    const float* WattS   = (const float*)d_in[14];
    const float* WattV   = (const float*)d_in[15];
    const float* Wread   = (const float*)d_in[16];
    const float* Winit   = (const float*)d_in[17];
    const float* bInit   = (const float*)d_in[18];
    float* out = (float*)d_out;

    // workspace carve-up (256B aligned)
    char* ws = (char*)d_ws;
    size_t off = 0;
    auto alloc = [&](size_t bytes) -> void* {
        void* p = ws + off;
        off += (bytes + 255) & ~(size_t)255;
        return p;
    };
    _Float16* srcX   = (_Float16*)alloc((size_t)Sn * Bn * En * 2);
    _Float16* tgtX   = (_Float16*)alloc((size_t)Tn * Bn * En * 2);
    _Float16* hWihF  = (_Float16*)alloc((size_t)2048 * 512 * 2);
    _Float16* hWhhF  = (_Float16*)alloc((size_t)2048 * 512 * 2);
    _Float16* hWihB  = (_Float16*)alloc((size_t)2048 * 512 * 2);
    _Float16* hWhhB  = (_Float16*)alloc((size_t)2048 * 512 * 2);
    _Float16* hWcomb = (_Float16*)alloc((size_t)2048 * 1536 * 2);
    _Float16* hWattS = (_Float16*)alloc((size_t)512 * 1024 * 2);
    _Float16* hWattV = (_Float16*)alloc((size_t)512 * 1536 * 2);
    _Float16* hWinit = (_Float16*)alloc((size_t)512 * 1024 * 2);
    _Float16* hWread = (_Float16*)alloc((size_t)Vt * 512 * 2);
    float*    XgF    = (float*)alloc((size_t)Sn * Bn * 2048 * 4);
    float*    XgB    = (float*)alloc((size_t)Sn * Bn * 2048 * 4);
    float*    outF   = (float*)alloc((size_t)Sn * Bn * Hn * 4);
    float*    outB   = (float*)alloc((size_t)Sn * Bn * Hn * 4);
    float*    hF     = (float*)alloc((size_t)Bn * Hn * 4);
    float*    cF     = (float*)alloc((size_t)Bn * Hn * 4);
    float*    hB     = (float*)alloc((size_t)Bn * Hn * 4);
    float*    cB     = (float*)alloc((size_t)Bn * Hn * 4);
    _Float16* hFh    = (_Float16*)alloc((size_t)Bn * Hn * 2);
    _Float16* hBh    = (_Float16*)alloc((size_t)Bn * Hn * 2);
    float*    Hg     = (float*)alloc((size_t)Bn * 2048 * 4);
    float*    srcEnc = (float*)alloc((size_t)Bn * Sn * 1024 * 4);
    _Float16* srcEncH= (_Float16*)alloc((size_t)Bn * Sn * 1024 * 2);
    float*    srcAtt = (float*)alloc((size_t)Bn * Sn * 512 * 4);
    _Float16* cinitH = (_Float16*)alloc((size_t)Bn * 1024 * 2);
    float*    cDec   = (float*)alloc((size_t)Bn * Hn * 4);
    float*    hDec   = (float*)alloc((size_t)Bn * Hn * 4);
    _Float16* hDecH  = (_Float16*)alloc((size_t)Bn * Hn * 2);
    _Float16* zeroAtt= (_Float16*)alloc((size_t)Bn * Hn * 2);
    _Float16* xcat   = (_Float16*)alloc((size_t)Bn * 1536 * 2);
    float*    gates  = (float*)alloc((size_t)Bn * 2048 * 4);
    _Float16* xav    = (_Float16*)alloc((size_t)Bn * 1536 * 2);
    _Float16* attVes = (_Float16*)alloc((size_t)Tn * Bn * Hn * 2);

    auto gemm = [&](const _Float16* A, const _Float16* W, const float* bias,
                    float* C, _Float16* Ch, int M, int N, int K, int act) {
        int tiles = (M / 16) * (N / 16);
        int blocks = (tiles + 7) / 8;
        k_gemm_f16<<<blocks, 256, 0, stream>>>(A, W, bias, C, Ch, M, N, K, act);
    };
    auto ew = [&](int n) { return dim3((n + 255) / 256); };

    // --- conversions / gathers --------------------------------------------
    k_embed_src<<<ew(Sn * Bn * En), 256, 0, stream>>>(srcTok, srcEmb, srcX);
    k_embed_tgt<<<ew(Tn * Bn * En), 256, 0, stream>>>(tgtTok, tgtEmb, tgtX);
    k_cvt_h<<<ew(2048 * 512), 256, 0, stream>>>(WihF, hWihF, 2048 * 512);
    k_cvt_h<<<ew(2048 * 512), 256, 0, stream>>>(WhhF, hWhhF, 2048 * 512);
    k_cvt_h<<<ew(2048 * 512), 256, 0, stream>>>(WihB, hWihB, 2048 * 512);
    k_cvt_h<<<ew(2048 * 512), 256, 0, stream>>>(WhhB, hWhhB, 2048 * 512);
    k_build_wcomb<<<ew(2048 * 1536), 256, 0, stream>>>(decWih, decWhh, hWcomb);
    k_cvt_h<<<ew(512 * 1024), 256, 0, stream>>>(WattS, hWattS, 512 * 1024);
    k_cvt_h<<<ew(512 * 1536), 256, 0, stream>>>(WattV, hWattV, 512 * 1536);
    k_cvt_h<<<ew(512 * 1024), 256, 0, stream>>>(Winit, hWinit, 512 * 1024);
    k_cvt_h<<<ew(Vt * 512), 256, 0, stream>>>(Wread, hWread, Vt * 512);
    k_zero_h<<<ew(Bn * Hn), 256, 0, stream>>>(zeroAtt, Bn * Hn);
    k_zero_f<<<1, 256, 0, stream>>>(out, Tn);

    // --- encoder input-gate precompute (big WMMA GEMMs) --------------------
    gemm(srcX, hWihF, nullptr, XgF, nullptr, Sn * Bn, 2048, 512, 0);
    gemm(srcX, hWihB, nullptr, XgB, nullptr, Sn * Bn, 2048, 512, 0);

    // --- forward encoder ----------------------------------------------------
    k_zero_f<<<ew(Bn * Hn), 256, 0, stream>>>(hF, Bn * Hn);
    k_zero_f<<<ew(Bn * Hn), 256, 0, stream>>>(cF, Bn * Hn);
    k_zero_h<<<ew(Bn * Hn), 256, 0, stream>>>(hFh, Bn * Hn);
    for (int t = 0; t < Sn; ++t) {
        gemm(hFh, hWhhF, nullptr, Hg, nullptr, Bn, 2048, 512, 0);
        k_enc_cell<<<ew(Bn * Hn), 256, 0, stream>>>(XgF, Hg, bF, lens, t, hF, cF, hFh, outF);
    }
    // --- backward encoder ---------------------------------------------------
    k_zero_f<<<ew(Bn * Hn), 256, 0, stream>>>(hB, Bn * Hn);
    k_zero_f<<<ew(Bn * Hn), 256, 0, stream>>>(cB, Bn * Hn);
    k_zero_h<<<ew(Bn * Hn), 256, 0, stream>>>(hBh, Bn * Hn);
    for (int tt = 0; tt < Sn; ++tt) {
        int t = Sn - 1 - tt;
        gemm(hBh, hWhhB, nullptr, Hg, nullptr, Bn, 2048, 512, 0);
        k_enc_cell<<<ew(Bn * Hn), 256, 0, stream>>>(XgB, Hg, bB, lens, t, hB, cB, hBh, outB);
    }

    // --- src_enc / src_att / decoder init -----------------------------------
    k_build_srcenc<<<ew(Bn * Sn * 1024), 256, 0, stream>>>(outF, outB, srcEnc, srcEncH);
    gemm(srcEncH, hWattS, nullptr, srcAtt, nullptr, Bn * Sn, 512, 1024, 0);
    k_build_cinit<<<ew(Bn * 1024), 256, 0, stream>>>(cF, cB, cinitH);
    gemm(cinitH, hWinit, bInit, cDec, nullptr, Bn, 512, 1024, 0);
    k_h0<<<ew(Bn * Hn), 256, 0, stream>>>(cDec, hDec, hDecH);

    // --- decoder loop --------------------------------------------------------
    for (int t = 0; t < Tn; ++t) {
        const _Float16* attPrev = (t == 0) ? zeroAtt : (attVes + (size_t)(t - 1) * Bn * Hn);
        k_build_xcat<<<ew(Bn * 1536), 256, 0, stream>>>(tgtX, t, attPrev, hDecH, xcat);
        gemm(xcat, hWcomb, decB, gates, nullptr, Bn, 2048, 1536, 0);
        k_dec_cell<<<ew(Bn * Hn), 256, 0, stream>>>(gates, hDec, cDec, hDecH);
        k_attn<<<Bn, 256, 0, stream>>>(srcAtt, srcEnc, hDec, hDecH, lens, xav);
        gemm(xav, hWattV, nullptr, nullptr, attVes + (size_t)t * Bn * Hn, Bn, 512, 1536, 1);
    }

    // --- fused readout: WMMA logits + online logsumexp + gold gather --------
    k_readout<<<(Tn * Bn) / 16, 256, 0, stream>>>(attVes, hWread, tgtTok, out);
}
